// AGCN_38113539785411
// MI455X (gfx1250) — matrix operand order repaired and checked by reading, not compiled, not run on hardware
//
#include <hip/hip_runtime.h>
#include <hip/hip_bf16.h>

typedef _Float16 half_t;
typedef __attribute__((ext_vector_type(16))) _Float16 v16h;
typedef __attribute__((ext_vector_type(8)))  _Float16 v8h;
typedef __attribute__((ext_vector_type(8)))  float    v8f;

#define CDIV(a,b) (((a)+(b)-1)/(b))

// ---------------------------------------------------------------------------
// kNN: one thread per (b,n); brute force over N candidates, keep 20 smallest
// squared distances (insertion keeps lower index on ties, matching top_k).
// ---------------------------------------------------------------------------
__global__ void knn_kernel(const float* __restrict__ pos, int Bc, int N,
                           int* __restrict__ idx, float* __restrict__ dsum)
{
    int t = blockIdx.x * blockDim.x + threadIdx.x;
    if (t >= Bc * N) return;
    int b = t / N, n = t % N;
    const float* pb = pos + (size_t)b * 3 * N;
    float px = pb[n], py = pb[N + n], pz = pb[2 * N + n];
    float bd[20]; int bi[20];
    #pragma unroll
    for (int i = 0; i < 20; i++) { bd[i] = 3.4e38f; bi[i] = 0; }
    for (int m = 0; m < N; m++) {
        float dx = pb[m] - px, dy = pb[N + m] - py, dz = pb[2 * N + m] - pz;
        float d2 = dx * dx + dy * dy + dz * dz;
        if (d2 < bd[19]) {
            int j = 19;
            while (j > 0 && bd[j - 1] > d2) { bd[j] = bd[j - 1]; bi[j] = bi[j - 1]; j--; }
            bd[j] = d2; bi[j] = m;
        }
    }
    float s = 0.f;
    for (int i = 0; i < 20; i++) { s += sqrtf(fmaxf(bd[i], 0.f)); idx[(size_t)t * 20 + i] = bi[i]; }
    dsum[t] = s;
}

// ---------------------------------------------------------------------------
// Build edge-feature matrix F (fp16): row r=(b,n,k), cols [xj-xi | xi | dp | dn]
// padded with zeros to Kpad (multiple of 32).
// ---------------------------------------------------------------------------
__global__ void build_feat_kernel(const float* __restrict__ x, const float* __restrict__ pos,
                                  const int* __restrict__ idx, half_t* __restrict__ F,
                                  int Bc, int N, int C, int Kpad)
{
    size_t t = (size_t)blockIdx.x * blockDim.x + threadIdx.x;
    size_t total = (size_t)Bc * N * 20 * Kpad;
    if (t >= total) return;
    int c = (int)(t % Kpad);
    size_t r = t / Kpad;
    size_t bn = r / 20;
    int n = (int)(bn % N);
    int b = (int)(bn / N);
    int j = idx[r];
    int twoC = 2 * C;
    float v = 0.f;
    if (c < C) {
        const float* xb = x + ((size_t)b * C + c) * N;
        v = xb[j] - xb[n];
    } else if (c < twoC) {
        v = x[((size_t)b * C + (c - C)) * N + n];
    } else if (c < twoC + 4) {
        const float* pb = pos + (size_t)b * 3 * N;
        float dx = pb[j] - pb[n], dy = pb[N + j] - pb[N + n], dz = pb[2 * N + j] - pb[2 * N + n];
        int g = c - twoC;
        if (g == 0) v = dx; else if (g == 1) v = dy; else if (g == 2) v = dz;
        else v = sqrtf(dx * dx + dy * dy + dz * dz + 1e-12f);
    }
    F[t] = (half_t)v;
}

// Cast [B,C,N] node features into F rows (b*N+n) x Kpad (for 1x1 convs).
__global__ void cast_xT_kernel(const float* __restrict__ x, half_t* __restrict__ F,
                               int Bc, int N, int C, int Kpad)
{
    size_t t = (size_t)blockIdx.x * blockDim.x + threadIdx.x;
    size_t total = (size_t)Bc * N * Kpad;
    if (t >= total) return;
    int c = (int)(t % Kpad);
    size_t r = t / Kpad;
    int n = (int)(r % N);
    int b = (int)(r / N);
    float v = (c < C) ? x[((size_t)b * C + c) * N + n] : 0.f;
    F[t] = (half_t)v;
}

// Pack [Wf | Wg] (and bias) into fp16 padded (Opad x Kpad) row-major.
__global__ void pack_weights_kernel(const float* __restrict__ Wf, const float* __restrict__ Wg,
                                    const float* __restrict__ bias,
                                    half_t* __restrict__ Wpad, float* __restrict__ biasPad,
                                    int O, int twoC, int Kpad, int Opad)
{
    int t = blockIdx.x * blockDim.x + threadIdx.x;
    if (t >= Opad * Kpad) return;
    int o = t / Kpad, c = t % Kpad;
    float v = 0.f;
    if (o < O) {
        if (c < twoC)            v = Wf[(size_t)o * twoC + c];
        else if (c < twoC + 4)   v = Wg[(size_t)o * 4 + (c - twoC)];
    }
    Wpad[t] = (half_t)v;
    if (c == 0) biasPad[o] = (o < O) ? bias[o] : 0.f;
}

// ---------------------------------------------------------------------------
// WMMA GEMM, register-blocked 16x64 per wave:
// H[R x Opad] = F[R x Kpad] * W[Opad x Kpad]^T + bias
// Each wave owns one 16-row strip and four 16-col tiles (Opad % 64 == 0);
// one A-tile load feeds four v_wmma_f32_16x16x32_f16 per 32-wide K step,
// cutting F re-read traffic 4x vs one-tile-per-wave (F >> L2 for big layers).
// A lane layout: row = lane%16, 16 contiguous K halves at (lane<16?0:16)
// (two 8-half loads). B lane layout: col = lane%16, 16 contiguous K halves at
// (lane<16?0:16). D: row = r + (lane<16?0:8), col = lane%16.
// ---------------------------------------------------------------------------
__global__ void wmma_gemm_kernel(const half_t* __restrict__ F, const half_t* __restrict__ W,
                                 const float* __restrict__ biasPad, float* __restrict__ H,
                                 int R, int Kpad, int Opad)
{
    int wid  = (blockIdx.x * blockDim.x + threadIdx.x) >> 5;
    int lane = threadIdx.x & 31;
    int blksO = Opad >> 6;               // 64-column blocks
    int tileR = wid / blksO;
    int blkO  = wid - tileR * blksO;
    if (tileR * 16 >= R) return;
    int hi  = lane >> 4;                 // 0 or 1
    int l15 = lane & 15;
    int rowA = tileR * 16 + l15;
    int col0 = blkO * 64 + l15;
    const half_t* Fp = F + (size_t)rowA * Kpad + (hi ? 16 : 0);
    const half_t* Wp = W + (size_t)col0 * Kpad + (hi ? 16 : 0);
    size_t wstride = (size_t)16 * Kpad;  // between adjacent 16-col tiles
    v8f acc[4];
    #pragma unroll
    for (int j = 0; j < 4; j++) {
        float bv = biasPad[col0 + 16 * j];
        #pragma unroll
        for (int i = 0; i < 8; i++) acc[j][i] = bv;
    }
    int steps = Kpad >> 5;
    for (int s = 0; s < steps; s++) {
        v8h a0 = *(const v8h*)(Fp);
        v8h a1 = *(const v8h*)(Fp + 8);
        v16h a;
        #pragma unroll
        for (int i = 0; i < 8; i++) { a[i] = a0[i]; a[i + 8] = a1[i]; }
        #pragma unroll
        for (int j = 0; j < 4; j++) {
            v16h bm = *(const v16h*)(Wp + (size_t)j * wstride);
            acc[j] = __builtin_amdgcn_wmma_f32_16x16x32_f16(false, a, false, bm,
                                                            (short)0, acc[j], false, false);
        }
        Fp += 32; Wp += 32;
    }
    int rbase = tileR * 16 + (hi ? 8 : 0);
    #pragma unroll
    for (int j = 0; j < 4; j++) {
        float* Ho = H + (size_t)rbase * Opad + col0 + 16 * j;
        #pragma unroll
        for (int i = 0; i < 8; i++) Ho[(size_t)i * Opad] = acc[j][i];
    }
}

// ---------------------------------------------------------------------------
// GroupNorm stats over (channels-in-group, N, K): one block per (b,g).
// H layout: ((b*N+n)*K + kk)*Opad + o
// ---------------------------------------------------------------------------
__global__ void gn_reduce_kernel(const float* __restrict__ H, float* __restrict__ stats,
                                 int Bc, int N, int K, int O, int Opad, int G)
{
    int b = blockIdx.x / G, g = blockIdx.x % G;
    int Og = O / G;
    size_t NK = (size_t)N * K;
    size_t cnt = (size_t)Og * NK;
    float s = 0.f, s2 = 0.f;
    for (size_t i = threadIdx.x; i < cnt; i += blockDim.x) {
        int o = g * Og + (int)(i / NK);
        size_t rem = i % NK;
        float v = H[((size_t)b * NK + rem) * Opad + o];
        s += v; s2 += v * v;
    }
    __shared__ float sh[256], sh2[256];
    sh[threadIdx.x] = s; sh2[threadIdx.x] = s2;
    __syncthreads();
    for (int off = blockDim.x >> 1; off > 0; off >>= 1) {
        if ((int)threadIdx.x < off) {
            sh[threadIdx.x]  += sh[threadIdx.x + off];
            sh2[threadIdx.x] += sh2[threadIdx.x + off];
        }
        __syncthreads();
    }
    if (threadIdx.x == 0) {
        float mu  = sh[0] / (float)cnt;
        float var = sh2[0] / (float)cnt - mu * mu;
        stats[blockIdx.x * 2]     = mu;
        stats[blockIdx.x * 2 + 1] = rsqrtf(var + 1e-5f);
    }
}

// Normalize + affine + leaky(0.2) + max over K; out: [B,O,N]
__global__ void gn_act_max_kernel(const float* __restrict__ H, const float* __restrict__ stats,
                                  const float* __restrict__ gamma, const float* __restrict__ beta,
                                  float* __restrict__ out,
                                  int Bc, int N, int K, int O, int Opad, int G)
{
    size_t t = (size_t)blockIdx.x * blockDim.x + threadIdx.x;
    size_t total = (size_t)Bc * O * N;
    if (t >= total) return;
    int n = (int)(t % N);
    int o = (int)((t / N) % O);
    int b = (int)(t / ((size_t)N * O));
    int g = o / (O / G);
    float mu = stats[(b * G + g) * 2], rs = stats[(b * G + g) * 2 + 1];
    float ga = gamma[o], be = beta[o];
    float m = -3.4e38f;
    size_t base = (size_t)(b * N + n) * K * Opad + o;
    for (int kk = 0; kk < K; kk++) {
        float v = H[base + (size_t)kk * Opad];
        v = (v - mu) * rs * ga + be;
        v = v >= 0.f ? v : 0.2f * v;
        m = fmaxf(m, v);
    }
    out[((size_t)b * O + o) * N + n] = m;
}

// top-n selection by rank (descending, lower index first on ties)
__global__ void topn_sel_kernel(const float* __restrict__ ds, int* __restrict__ sel,
                                int Bc, int N, int nsel)
{
    int t = blockIdx.x * blockDim.x + threadIdx.x;
    if (t >= Bc * N) return;
    int b = t / N, i = t % N;
    float v = ds[b * N + i];
    int rank = 0;
    for (int j = 0; j < N; j++) {
        float u = ds[b * N + j];
        if (u > v || (u == v && j < i)) rank++;
    }
    if (rank < nsel) sel[b * nsel + rank] = i;
}

__global__ void gather_ch_kernel(const float* __restrict__ src, const int* __restrict__ sel,
                                 float* __restrict__ dst, int Bc, int C, int N, int nsel)
{
    size_t t = (size_t)blockIdx.x * blockDim.x + threadIdx.x;
    size_t total = (size_t)Bc * C * nsel;
    if (t >= total) return;
    int j = (int)(t % nsel);
    int c = (int)((t / nsel) % C);
    int b = (int)(t / ((size_t)nsel * C));
    dst[t] = src[((size_t)b * C + c) * N + sel[b * nsel + j]];
}

// 3-NN search for interpolation: normalized weights stored
__global__ void interp_nn3_kernel(const float* __restrict__ posy, const float* __restrict__ posx,
                                  int* __restrict__ widx, float* __restrict__ w,
                                  int Bc, int Ny, int Nx)
{
    int t = blockIdx.x * blockDim.x + threadIdx.x;
    if (t >= Bc * Ny) return;
    int b = t / Ny, n = t % Ny;
    const float* py = posy + (size_t)b * 3 * Ny;
    const float* px = posx + (size_t)b * 3 * Nx;
    float ax = py[n], ay = py[Ny + n], az = py[2 * Ny + n];
    float bd[3] = {3.4e38f, 3.4e38f, 3.4e38f};
    int   bi[3] = {0, 0, 0};
    for (int m = 0; m < Nx; m++) {
        float dx = px[m] - ax, dy = px[Nx + m] - ay, dz = px[2 * Nx + m] - az;
        float d2 = dx * dx + dy * dy + dz * dz;
        if (d2 < bd[2]) {
            int j = 2;
            while (j > 0 && bd[j - 1] > d2) { bd[j] = bd[j - 1]; bi[j] = bi[j - 1]; j--; }
            bd[j] = d2; bi[j] = m;
        }
    }
    float ws = 0.f, wv[3];
    #pragma unroll
    for (int i = 0; i < 3; i++) { wv[i] = 1.f / (fmaxf(bd[i], 0.f) + 1e-8f); ws += wv[i]; }
    #pragma unroll
    for (int i = 0; i < 3; i++) { widx[(size_t)t * 3 + i] = bi[i]; w[(size_t)t * 3 + i] = wv[i] / ws; }
}

// Apply interpolation, writing into channel slice [coff, coff+C) of dst [B,Ctot,Ny]
__global__ void interp_apply_kernel(const float* __restrict__ fx, const int* __restrict__ widx,
                                    const float* __restrict__ w, float* __restrict__ out,
                                    int Bc, int C, int Ny, int Nx, int Ctot, int coff)
{
    size_t t = (size_t)blockIdx.x * blockDim.x + threadIdx.x;
    size_t total = (size_t)Bc * C * Ny;
    if (t >= total) return;
    int n = (int)(t % Ny);
    int c = (int)((t / Ny) % C);
    int b = (int)(t / ((size_t)Ny * C));
    const int*   wi = widx + ((size_t)b * Ny + n) * 3;
    const float* wf = w    + ((size_t)b * Ny + n) * 3;
    const float* f  = fx   + ((size_t)b * C + c) * Nx;
    float s = wf[0] * f[wi[0]] + wf[1] * f[wi[1]] + wf[2] * f[wi[2]];
    out[((size_t)b * Ctot + coff + c) * Ny + n] = s;
}

__global__ void copy_ch_kernel(const float* __restrict__ src, float* __restrict__ dst,
                               int Bc, int C, int N, int Ctot, int coff)
{
    size_t t = (size_t)blockIdx.x * blockDim.x + threadIdx.x;
    size_t total = (size_t)Bc * C * N;
    if (t >= total) return;
    int n = (int)(t % N);
    int c = (int)((t / N) % C);
    int b = (int)(t / ((size_t)N * C));
    dst[((size_t)b * Ctot + coff + c) * N + n] = src[((size_t)b * C + c) * N + n];
}

// H[(b*N+n)*Opad + o] -> out[(b*O+o)*N + n]  (final logits)
__global__ void h_to_out_kernel(const float* __restrict__ H, float* __restrict__ out,
                                int Bc, int N, int O, int Opad)
{
    size_t t = (size_t)blockIdx.x * blockDim.x + threadIdx.x;
    size_t total = (size_t)Bc * O * N;
    if (t >= total) return;
    int n = (int)(t % N);
    int o = (int)((t / N) % O);
    int b = (int)(t / ((size_t)N * O));
    out[((size_t)b * O + o) * N + n] = H[((size_t)b * N + n) * Opad + o];
}

// ---------------------------------------------------------------------------
// Host orchestration
// ---------------------------------------------------------------------------
static void launch_gemm(const half_t* F, const half_t* W, const float* bias, float* H,
                        int R, int Kpad, int Opad, hipStream_t stream)
{
    size_t waves = ((size_t)R / 16) * (Opad / 64);
    size_t th = waves * 32;
    wmma_gemm_kernel<<<(unsigned)CDIV(th, 256), 256, 0, stream>>>(F, W, bias, H, R, Kpad, Opad);
}

struct Scratch {
    char* base; size_t off;
    void* get(size_t bytes) { void* p = base + off; off += (bytes + 255) & ~(size_t)255; return p; }
};

static void edge_conv(const float* x, const float* pos, const int* idx,
                      const float* Wf, const float* Wg, const float* bias,
                      const float* gamma, const float* beta, float* out,
                      half_t* Fbuf, half_t* Wbuf, float* biasbuf, float* Hbuf, float* stats,
                      int N, int C, int O, hipStream_t stream)
{
    const int Bc = 4, K = 20, G = 4;
    int twoC = 2 * C;
    int Kpad = CDIV(twoC + 4, 32) * 32;
    int Opad = CDIV(O, 64) * 64;        // multiple of 64 for the blocked GEMM
    int R = Bc * N * K;                 // always a multiple of 16 here
    size_t totF = (size_t)R * Kpad;
    build_feat_kernel<<<(unsigned)CDIV(totF, 256), 256, 0, stream>>>(x, pos, idx, Fbuf, Bc, N, C, Kpad);
    int totW = Opad * Kpad;
    pack_weights_kernel<<<(unsigned)CDIV(totW, 256), 256, 0, stream>>>(Wf, Wg, bias, Wbuf, biasbuf, O, twoC, Kpad, Opad);
    launch_gemm(Fbuf, Wbuf, biasbuf, Hbuf, R, Kpad, Opad, stream);
    gn_reduce_kernel<<<Bc * G, 256, 0, stream>>>(Hbuf, stats, Bc, N, K, O, Opad, G);
    size_t totA = (size_t)Bc * O * N;
    gn_act_max_kernel<<<(unsigned)CDIV(totA, 256), 256, 0, stream>>>(Hbuf, stats, gamma, beta, out, Bc, N, K, O, Opad, G);
}

extern "C" void kernel_launch(void* const* d_in, const int* in_sizes, int n_in,
                              void* d_out, int out_size, void* d_ws, size_t ws_size,
                              hipStream_t stream)
{
    (void)in_sizes; (void)n_in; (void)out_size; (void)ws_size;
    const int Bc = 4, N = 2048, K = 20, SEG = 50;
    const float* x = (const float*)d_in[0];
    // params: e0..e6 => {Wf,Wg,b,gamma,beta} at 1+5*i; then c4_W..c5_b
    auto EP = [&](int layer, int j) { return (const float*)d_in[1 + 5 * layer + j]; };
    const float* c4_W = (const float*)d_in[36];
    const float* c4_b = (const float*)d_in[37];
    const float* c4_g = (const float*)d_in[38];
    const float* c4_be = (const float*)d_in[39];
    const float* c5_W = (const float*)d_in[40];
    const float* c5_b = (const float*)d_in[41];

    Scratch ws{(char*)d_ws, 0};
    int*   idxM = (int*)  ws.get((size_t)Bc * N * K * 4);
    float* dsM  = (float*)ws.get((size_t)Bc * N * 4);
    int*   idx2 = (int*)  ws.get((size_t)Bc * 512 * K * 4);
    float* ds2  = (float*)ws.get((size_t)Bc * 512 * 4);
    int*   idx3 = (int*)  ws.get((size_t)Bc * 256 * K * 4);
    float* ds3  = (float*)ws.get((size_t)Bc * 256 * 4);
    half_t* Fbuf = (half_t*)ws.get((size_t)Bc * N * K * 800 * 2);   // max Kpad = 800 (e5)
    half_t* Wbuf = (half_t*)ws.get((size_t)1 << 20);                // max 803 KB (e4)
    float* biasbuf = (float*)ws.get(512 * 4);
    float* Hbuf  = (float*)ws.get((size_t)Bc * N * K * 256 * 4);    // max Opad = 256 @ N=2048
    float* stats = (float*)ws.get(Bc * 4 * 2 * 4);
    float* xa   = (float*)ws.get((size_t)Bc * 64  * N   * 4);
    float* x1   = (float*)ws.get((size_t)Bc * 128 * N   * 4);
    float* x2in = (float*)ws.get((size_t)Bc * 128 * 512 * 4);
    float* pos2 = (float*)ws.get((size_t)Bc * 3   * 512 * 4);
    float* x2   = (float*)ws.get((size_t)Bc * 256 * 512 * 4);
    float* x3in = (float*)ws.get((size_t)Bc * 256 * 256 * 4);
    float* pos3 = (float*)ws.get((size_t)Bc * 3   * 256 * 4);
    float* x3   = (float*)ws.get((size_t)Bc * 512 * 256 * 4);
    float* x4in = (float*)ws.get((size_t)Bc * 768 * 512 * 4);
    float* x4   = (float*)ws.get((size_t)Bc * 256 * 512 * 4);
    float* x5in = (float*)ws.get((size_t)Bc * 384 * N   * 4);
    float* x5   = (float*)ws.get((size_t)Bc * 256 * N   * 4);
    float* x6   = (float*)ws.get((size_t)Bc * 256 * N   * 4);
    float* c4o  = (float*)ws.get((size_t)Bc * 128 * N   * 4);
    int*   widx = (int*)  ws.get((size_t)Bc * N * 3 * 4);
    float* wgt  = (float*)ws.get((size_t)Bc * N * 3 * 4);
    int*   sel  = (int*)  ws.get((size_t)Bc * 512 * 4);

    // ---- kNN on full cloud (shared by e0,e1,e5,e6) ----
    knn_kernel<<<CDIV(Bc * N, 256), 256, 0, stream>>>(x, Bc, N, idxM, dsM);
    // ---- e0, e1 ----
    edge_conv(x,  x, idxM, EP(0,0), EP(0,1), EP(0,2), EP(0,3), EP(0,4), xa,
              Fbuf, Wbuf, biasbuf, Hbuf, stats, N, 3, 64, stream);
    edge_conv(xa, x, idxM, EP(1,0), EP(1,1), EP(1,2), EP(1,3), EP(1,4), x1,
              Fbuf, Wbuf, biasbuf, Hbuf, stats, N, 64, 128, stream);
    // ---- idis -> 512 ----
    topn_sel_kernel<<<CDIV(Bc * N, 256), 256, 0, stream>>>(dsM, sel, Bc, N, 512);
    gather_ch_kernel<<<CDIV(Bc * 3 * 512, 256), 256, 0, stream>>>(x,  sel, pos2, Bc, 3,   N, 512);
    gather_ch_kernel<<<CDIV(Bc * 128 * 512, 256), 256, 0, stream>>>(x1, sel, x2in, Bc, 128, N, 512);
    // ---- e2 ----
    knn_kernel<<<CDIV(Bc * 512, 256), 256, 0, stream>>>(pos2, Bc, 512, idx2, ds2);
    edge_conv(x2in, pos2, idx2, EP(2,0), EP(2,1), EP(2,2), EP(2,3), EP(2,4), x2,
              Fbuf, Wbuf, biasbuf, Hbuf, stats, 512, 128, 256, stream);
    // ---- idis -> 256 ----
    topn_sel_kernel<<<CDIV(Bc * 512, 256), 256, 0, stream>>>(ds2, sel, Bc, 512, 256);
    gather_ch_kernel<<<CDIV(Bc * 3 * 256, 256), 256, 0, stream>>>(pos2, sel, pos3, Bc, 3,   512, 256);
    gather_ch_kernel<<<CDIV(Bc * 256 * 256, 256), 256, 0, stream>>>(x2,  sel, x3in, Bc, 256, 512, 256);
    // ---- e3 ----
    knn_kernel<<<CDIV(Bc * 256, 256), 256, 0, stream>>>(pos3, Bc, 256, idx3, ds3);
    edge_conv(x3in, pos3, idx3, EP(3,0), EP(3,1), EP(3,2), EP(3,3), EP(3,4), x3,
              Fbuf, Wbuf, biasbuf, Hbuf, stats, 256, 256, 512, stream);
    // ---- interpolate x3 -> 512 pts, concat x2 ----
    interp_nn3_kernel<<<CDIV(Bc * 512, 256), 256, 0, stream>>>(pos2, pos3, widx, wgt, Bc, 512, 256);
    interp_apply_kernel<<<CDIV(Bc * 512 * 512, 256), 256, 0, stream>>>(x3, widx, wgt, x4in, Bc, 512, 512, 256, 768, 0);
    copy_ch_kernel<<<CDIV(Bc * 256 * 512, 256), 256, 0, stream>>>(x2, x4in, Bc, 256, 512, 768, 512);
    // ---- e4 ----
    edge_conv(x4in, pos2, idx2, EP(4,0), EP(4,1), EP(4,2), EP(4,3), EP(4,4), x4,
              Fbuf, Wbuf, biasbuf, Hbuf, stats, 512, 768, 256, stream);
    // ---- interpolate x4 -> 2048 pts, concat x1 ----
    interp_nn3_kernel<<<CDIV(Bc * N, 256), 256, 0, stream>>>(x, pos2, widx, wgt, Bc, N, 512);
    interp_apply_kernel<<<(unsigned)CDIV((size_t)Bc * 256 * N, 256), 256, 0, stream>>>(x4, widx, wgt, x5in, Bc, 256, N, 512, 384, 0);
    copy_ch_kernel<<<(unsigned)CDIV((size_t)Bc * 128 * N, 256), 256, 0, stream>>>(x1, x5in, Bc, 128, N, 384, 256);
    // ---- e5, e6 ----
    edge_conv(x5in, x, idxM, EP(5,0), EP(5,1), EP(5,2), EP(5,3), EP(5,4), x5,
              Fbuf, Wbuf, biasbuf, Hbuf, stats, N, 384, 256, stream);
    edge_conv(x5, x, idxM, EP(6,0), EP(6,1), EP(6,2), EP(6,3), EP(6,4), x6,
              Fbuf, Wbuf, biasbuf, Hbuf, stats, N, 256, 256, stream);
    // ---- c4: 1x1 conv (WMMA) + GN + leaky ----
    {
        int Cin = 256, O = 128, Kpad = 256, Opad = 128, R = Bc * N;
        cast_xT_kernel<<<(unsigned)CDIV((size_t)R * Kpad, 256), 256, 0, stream>>>(x6, Fbuf, Bc, N, Cin, Kpad);
        pack_weights_kernel<<<CDIV(Opad * Kpad, 256), 256, 0, stream>>>(c4_W, c4_W, c4_b, Wbuf, biasbuf, O, Cin, Kpad, Opad);
        launch_gemm(Fbuf, Wbuf, biasbuf, Hbuf, R, Kpad, Opad, stream);
        gn_reduce_kernel<<<Bc * 4, 256, 0, stream>>>(Hbuf, stats, Bc, N, 1, O, Opad, 4);
        gn_act_max_kernel<<<(unsigned)CDIV((size_t)Bc * O * N, 256), 256, 0, stream>>>(Hbuf, stats, c4_g, c4_be, c4o, Bc, N, 1, O, Opad, 4);
    }
    // ---- c5: final 1x1 conv (WMMA) ----
    {
        int Cin = 128, O = SEG, Kpad = 128, Opad = 64, R = Bc * N;
        cast_xT_kernel<<<(unsigned)CDIV((size_t)R * Kpad, 256), 256, 0, stream>>>(c4o, Fbuf, Bc, N, Cin, Kpad);
        pack_weights_kernel<<<CDIV(Opad * Kpad, 256), 256, 0, stream>>>(c5_W, c5_W, c5_b, Wbuf, biasbuf, O, Cin, Kpad, Opad);
        launch_gemm(Fbuf, Wbuf, biasbuf, Hbuf, R, Kpad, Opad, stream);
        h_to_out_kernel<<<(unsigned)CDIV((size_t)Bc * O * N, 256), 256, 0, stream>>>(Hbuf, (float*)d_out, Bc, N, O, Opad);
    }
}
